// MambaBlock_2911987827241
// MI455X (gfx1250) — matrix operand order repaired and checked by reading, not compiled, not run on hardware
//
#include <hip/hip_runtime.h>
#include <hip/hip_bf16.h>
#include <math.h>

// ---------------- problem constants ----------------
#define D_MODEL 1024
#define D_INNER 2048
#define D_STATE 16
#define DT_RANK 64
#define D_CONV  4
#define BATCH   2
#define SEQ     1024
#define M_ROWS  (BATCH * SEQ)      // 2048 flattened tokens

// ---------------- WMMA vector types ----------------
typedef __attribute__((ext_vector_type(16))) __bf16 v16bf;
typedef __attribute__((ext_vector_type(8)))  __bf16 v8bf;
typedef __attribute__((ext_vector_type(4)))  __bf16 v4bf;
typedef __attribute__((ext_vector_type(8)))  float  v8f;

// ---------------- GEMM tiling ----------------
// Block tile 128x128 (8 waves, each wave owns 32x64 = 2x4 WMMA tiles).
// Requirements (not checked; the launcher's fixed shapes meet them): M % 128 == 0, K % 32 == 0.
// GUARD handles only the N dimension (needed for the N=96 x_dbl GEMM).
#define TM 128
#define TN 128
#define TK 32
#define LDA_P 40   // padded LDS row (bf16 elems); row stride 80B (8B-aligned)
#define LDB_P 40

// C[M,N] = A[M,K] @ B[K,N]  (f32 in/out, bf16 WMMA core, f32 accumulate)
// EPI: 0 = none, 1 = softplus(v + bias[col])
template<bool GUARD, int EPI>
__global__ __launch_bounds__(256)
void gemm_bf16_wmma(const float* __restrict__ A, int lda,
                    const float* __restrict__ Bw, int ldb,
                    float* __restrict__ C, int ldc,
                    int M, int N, int K,
                    const float* __restrict__ bias)
{
    __shared__ __bf16 lA[2][TM * LDA_P];   // [m][k]
    __shared__ __bf16 lB[2][TN * LDB_P];   // transposed: [n][k]

    const int tid   = threadIdx.x;
    const int lane  = tid & 31;
    const int wave  = tid >> 5;
    const int wm    = (wave & 3) * 32;   // wave row block within tile
    const int wn    = (wave >> 2) * 64;  // wave col block within tile
    const int lhalf = lane >> 4;         // 0 = lanes 0-15, 1 = lanes 16-31
    const int l16   = lane & 15;

    const int m0 = blockIdx.y * TM;
    const int n0 = blockIdx.x * TN;

    // register staging for the double-buffered pipeline
    float4 ar[4];   // A: 128x32 f32 = 1024 float4 slots / 256 threads (4 k per slot)
    float4 br[4];   // B: each slot = 4 consecutive k for one n (packs to one b64 store)

    auto fetchT = [&](int kt) {
        #pragma unroll
        for (int i = 0; i < 4; ++i) {
            int q = tid + (i << 8);
            int m = q >> 3, k = (q & 7) << 2;
            ar[i] = *(const float4*)(A + (size_t)(m0 + m) * lda + kt + k);
        }
        #pragma unroll
        for (int i = 0; i < 4; ++i) {
            int q  = tid + (i << 8);          // 1024 slots: 128 n x 8 k-groups
            int n  = q & 127;
            int kg = q >> 7;                  // 0..7, 4 k each
            float4 v = make_float4(0.f, 0.f, 0.f, 0.f);
            if (!GUARD || (n0 + n) < N) {
                const float* p = Bw + (size_t)(kt + (kg << 2)) * ldb + n0 + n;
                v.x = p[0];
                v.y = p[(size_t)ldb];
                v.z = p[(size_t)2 * ldb];
                v.w = p[(size_t)3 * ldb];
            }
            br[i] = v;
        }
    };
    auto stageT = [&](int buf) {
        #pragma unroll
        for (int i = 0; i < 4; ++i) {
            int q = tid + (i << 8);
            int m = q >> 3, k = (q & 7) << 2;
            v4bf p;
            p[0] = (__bf16)ar[i].x; p[1] = (__bf16)ar[i].y;
            p[2] = (__bf16)ar[i].z; p[3] = (__bf16)ar[i].w;
            *(v4bf*)&lA[buf][m * LDA_P + k] = p;
        }
        #pragma unroll
        for (int i = 0; i < 4; ++i) {
            int q  = tid + (i << 8);
            int n  = q & 127;
            int kg = q >> 7;
            v4bf p;
            p[0] = (__bf16)br[i].x; p[1] = (__bf16)br[i].y;
            p[2] = (__bf16)br[i].z; p[3] = (__bf16)br[i].w;
            *(v4bf*)&lB[buf][n * LDB_P + (kg << 2)] = p;   // one ds_store_b64
        }
    };

    v8f zero = {0.f,0.f,0.f,0.f,0.f,0.f,0.f,0.f};
    v8f acc[2][4];
    #pragma unroll
    for (int fm = 0; fm < 2; ++fm)
        #pragma unroll
        for (int fn = 0; fn < 4; ++fn) acc[fm][fn] = zero;

    // pipeline prologue: tile 0 into LDS buf0, tile 1 into registers
    fetchT(0);
    stageT(0);
    if (K > TK) fetchT(TK);
    __syncthreads();

    int buf = 0;
    for (int kt = 0; kt < K; kt += TK) {
        // ---- fragments per ISA 16-bit layouts ----
        // A 16x32: lane<16 holds row M=l16, K in {0..7}U{16..23}; lane>=16 shifted by 8
        // B 32x16 (LDS transposed): lane<16 holds col N=l16, K in {0..7}U{16..23}; lane>=16 shifted by 8
        v16bf af[2], bfr[4];
        #pragma unroll
        for (int f = 0; f < 2; ++f) {
            int row = wm + f * 16 + l16;
            v8bf a0 = *(const v8bf*)&lA[buf][row * LDA_P + lhalf * 8];
            v8bf a1 = *(const v8bf*)&lA[buf][row * LDA_P + 16 + lhalf * 8];
            #pragma unroll
            for (int e = 0; e < 8; ++e) { af[f][e] = a0[e]; af[f][8 + e] = a1[e]; }
        }
        #pragma unroll
        for (int f = 0; f < 4; ++f) {
            int col = wn + f * 16 + l16;
            v8bf b0 = *(const v8bf*)&lB[buf][col * LDB_P + lhalf * 8];
            v8bf b1 = *(const v8bf*)&lB[buf][col * LDB_P + 16 + lhalf * 8];
            #pragma unroll
            for (int e = 0; e < 8; ++e) { bfr[f][e] = b0[e]; bfr[f][8 + e] = b1[e]; }
        }

        #pragma unroll
        for (int fm = 0; fm < 2; ++fm)
            #pragma unroll
            for (int fn = 0; fn < 4; ++fn)
                acc[fm][fn] = __builtin_amdgcn_wmma_f32_16x16x32_bf16(
                    false, af[fm], false, bfr[fn],
                    (short)0, acc[fm][fn], false, false);

        if (kt + TK < K) {
            stageT(buf ^ 1);                          // regs(tile kt+1) -> other LDS buffer
            if (kt + 2 * TK < K) fetchT(kt + 2 * TK); // global fetch tile kt+2 -> regs
            __syncthreads();
            buf ^= 1;
        }
    }

    // ---- store: C/D layout: VGPR r, lane<16 -> M=r, lane>=16 -> M=8+r; N=l16 ----
    #pragma unroll
    for (int fm = 0; fm < 2; ++fm) {
        int rowb = m0 + wm + fm * 16 + lhalf * 8;
        #pragma unroll
        for (int fn = 0; fn < 4; ++fn) {
            int col = n0 + wn + fn * 16 + l16;
            if (GUARD && col >= N) continue;
            #pragma unroll
            for (int r = 0; r < 8; ++r) {
                float v = acc[fm][fn][r];
                if (EPI == 1) {
                    v += bias[col];
                    v = (v > 20.f) ? v : log1pf(__expf(v));   // softplus
                }
                C[(size_t)(rowb + r) * ldc + col] = v;
            }
        }
    }
}

// ---------------- depthwise causal conv(4) + bias + SiLU ----------------
// u lives in xz[:, 0:D_INNER]; writes u_conv[M_ROWS][D_INNER]
__global__ __launch_bounds__(256)
void conv_silu_kernel(const float* __restrict__ xz,
                      const float* __restrict__ conv_w,
                      const float* __restrict__ conv_b,
                      float* __restrict__ u_conv)
{
    int idx = blockIdx.x * 256 + threadIdx.x;       // B*L*D_INNER threads
    int d  = idx & (D_INNER - 1);
    int bl = idx >> 11;
    int l  = bl & (SEQ - 1);
    int b  = bl >> 10;

    float acc = conv_b[d];
    #pragma unroll
    for (int j = 0; j < D_CONV; ++j) {
        int ls = l - (D_CONV - 1) + j;
        if (ls >= 0)
            acc += xz[((size_t)(b * SEQ + ls)) * (2 * D_INNER) + d] * conv_w[d * D_CONV + j];
    }
    float s = acc / (1.f + __expf(-acc));           // SiLU
    u_conv[(size_t)bl * D_INNER + d] = s;
}

// ---------------- selective scan ----------------
// one thread per (b, d, n); 16-lane butterfly reduce over n; fuses +u*D and *silu(res)
__global__ __launch_bounds__(256)
void scan_kernel(const float* __restrict__ delta,
                 const float* __restrict__ u,
                 const float* __restrict__ xdbl,
                 const float* __restrict__ A_log,
                 const float* __restrict__ Dp,
                 const float* __restrict__ xz,
                 float* __restrict__ y)
{
    int idx = blockIdx.x * 256 + threadIdx.x;       // B * D_INNER * D_STATE = 65536
    int n = idx & (D_STATE - 1);
    int d = (idx >> 4) & (D_INNER - 1);
    int b = idx >> 15;

    float Ad = -__expf(A_log[d * D_STATE + n]);
    float Dd = Dp[d];
    float h  = 0.f;

    for (int l = 0; l < SEQ; ++l) {
        int rb = b * SEQ + l;
        float dl = delta[(size_t)rb * D_INNER + d];
        float uv = u[(size_t)rb * D_INNER + d];
        const float* xr = xdbl + (size_t)rb * (DT_RANK + 2 * D_STATE);
        float Bn = xr[DT_RANK + n];
        float Cn = xr[DT_RANK + D_STATE + n];

        h = __expf(dl * Ad) * h + (dl * uv) * Bn;   // h = dA*h + dBu
        float yv = h * Cn;

        // sum over n within each 16-lane group (wave32)
        yv += __shfl_xor(yv, 1, 32);
        yv += __shfl_xor(yv, 2, 32);
        yv += __shfl_xor(yv, 4, 32);
        yv += __shfl_xor(yv, 8, 32);

        if (n == 0) {
            yv += uv * Dd;
            float r = xz[(size_t)rb * (2 * D_INNER) + D_INNER + d];
            yv *= r / (1.f + __expf(-r));           // * silu(res)
            y[(size_t)rb * D_INNER + d] = yv;
        }
    }
}

// ---------------- launcher ----------------
extern "C" void kernel_launch(void* const* d_in, const int* in_sizes, int n_in,
                              void* d_out, int out_size, void* d_ws, size_t ws_size,
                              hipStream_t stream)
{
    const float* x      = (const float*)d_in[0];
    const float* W_in   = (const float*)d_in[1];
    const float* conv_w = (const float*)d_in[2];
    const float* conv_b = (const float*)d_in[3];
    const float* W_x    = (const float*)d_in[4];
    const float* W_dt   = (const float*)d_in[5];
    const float* b_dt   = (const float*)d_in[6];
    const float* A_log  = (const float*)d_in[7];
    const float* Dp     = (const float*)d_in[8];
    const float* W_out  = (const float*)d_in[9];
    float* out = (float*)d_out;

    // workspace layout
    float* xz    = (float*)d_ws;                                   // 2048 x 4096
    float* uconv = xz    + (size_t)M_ROWS * (2 * D_INNER);         // 2048 x 2048
    float* xdbl  = uconv + (size_t)M_ROWS * D_INNER;               // 2048 x 96
    float* delta = xdbl  + (size_t)M_ROWS * (DT_RANK + 2*D_STATE); // 2048 x 2048
    float* yscan = delta + (size_t)M_ROWS * D_INNER;               // 2048 x 2048

    dim3 blk(256);

    // 1) xz = x @ W_in          [2048 x 1024] @ [1024 x 4096]   (exact tiles)
    gemm_bf16_wmma<false, 0><<<dim3((2*D_INNER)/TN, M_ROWS/TM), blk, 0, stream>>>(
        x, D_MODEL, W_in, 2*D_INNER, xz, 2*D_INNER,
        M_ROWS, 2*D_INNER, D_MODEL, nullptr);

    // 2) depthwise conv + SiLU
    conv_silu_kernel<<<(BATCH * SEQ * D_INNER) / 256, blk, 0, stream>>>(
        xz, conv_w, conv_b, uconv);

    // 3) x_dbl = u @ W_x        [2048 x 2048] @ [2048 x 96]     (N=96: guarded)
    gemm_bf16_wmma<true, 0><<<dim3(1, M_ROWS/TM), blk, 0, stream>>>(
        uconv, D_INNER, W_x, DT_RANK + 2*D_STATE, xdbl, DT_RANK + 2*D_STATE,
        M_ROWS, DT_RANK + 2*D_STATE, D_INNER, nullptr);

    // 4) delta = softplus(dt_low @ W_dt + b_dt)   [2048 x 64] @ [64 x 2048]
    gemm_bf16_wmma<false, 1><<<dim3(D_INNER/TN, M_ROWS/TM), blk, 0, stream>>>(
        xdbl, DT_RANK + 2*D_STATE, W_dt, D_INNER, delta, D_INNER,
        M_ROWS, D_INNER, DT_RANK, b_dt);

    // 5) selective scan (fuses +u*D and *silu(res))
    scan_kernel<<<(BATCH * D_INNER * D_STATE) / 256, blk, 0, stream>>>(
        delta, uconv, xdbl, A_log, Dp, xz, yscan);

    // 6) out = y @ W_out        [2048 x 2048] @ [2048 x 1024]
    gemm_bf16_wmma<false, 0><<<dim3(D_MODEL/TN, M_ROWS/TM), blk, 0, stream>>>(
        yscan, D_INNER, W_out, D_MODEL, out, D_MODEL,
        M_ROWS, D_MODEL, D_INNER, nullptr);
}
